// CausalSelfAttention_28570122453437
// MI455X (gfx1250) — matrix operand order repaired and checked by reading, not compiled, not run on hardware
//
#include <hip/hip_runtime.h>
#include <stdint.h>

typedef _Float16 v4h  __attribute__((ext_vector_type(4)));
typedef _Float16 v8h  __attribute__((ext_vector_type(8)));
typedef _Float16 v16h __attribute__((ext_vector_type(16)));
typedef float    v4f  __attribute__((ext_vector_type(4)));
typedef float    v8f  __attribute__((ext_vector_type(8)));
typedef int      v4i  __attribute__((ext_vector_type(4)));

constexpr int kB = 2, kN = 2048, kD = 1024, kH = 16, kDH = 64, kE = 192;

union V16 { v16h v; v8h h[2]; };

__device__ inline v8f wmma_f16(v16h a, v16h b, v8f c) {
  return __builtin_amdgcn_wmma_f32_16x16x32_f16(false, a, false, b, (short)0, c,
                                                false, false);
}

#if __has_builtin(__builtin_amdgcn_global_load_async_to_lds_b128) && \
    __has_builtin(__builtin_amdgcn_s_wait_asynccnt)
#define USE_ASYNC 1
// Builtin signature (from clang diagnostic): arg0 is v4i in addrspace(1).
// Generic pointers: global VA == numeric value; LDS offset == low 32 bits.
#define AS1(p) ((__attribute__((address_space(1))) v4i*)(uintptr_t)(p))
#define AS3(p) ((__attribute__((address_space(3))) v4i*)(uint32_t)(uintptr_t)(p))
#else
#define USE_ASYNC 0
#endif

// ---------------------------------------------------------------------------
// Kernel 1: per-head KQV projection.  out[b,h,n,e] = x[b,n,:]·Wkqv[h,:,e]+bias
// WG: 4 waves, tile 128(M rows of n) x 64(N cols of e); each wave 32Mx64N
// (2 A-frags x 4 B-frags = 8 WMMA per K-step).  blockIdx.y in {0,1,2} selects
// the 64-wide e-block == the k/q/v section directly.
// ---------------------------------------------------------------------------
__global__ __launch_bounds__(128) void kqv_kernel(
    const float* __restrict__ x, const float* __restrict__ Wk,
    const float* __restrict__ bk, _Float16* __restrict__ kqvh) {
  __shared__ __align__(16) _Float16 Xs[128 * 32];
  __shared__ __align__(16) _Float16 Ws[64 * 32];  // transposed [n][k]
  const int tid = threadIdx.x;
  const int wid = tid >> 5, lane = tid & 31, lm = lane & 15;
  const int kb8 = (lane >> 4) * 8, kb16 = (lane >> 4) * 16;
  const int mbase = blockIdx.x * 128;
  const int sect  = blockIdx.y;  // 0=k, 1=q, 2=v
  const int bh    = blockIdx.z;
  const int b = bh / kH, h = bh % kH;

  const float* xrow = &x[((size_t)b * kN + mbase) * kD];
  const float* wrow = &Wk[(size_t)h * kD * kE + sect * 64];

  v8f acc[2][4] = {};
  for (int kk = 0; kk < kD / 32; ++kk) {
    const int k0 = kk * 32;
    // stage X 128x32 (f32 -> f16): phase 1 = clause of float4 loads,
    // phase 2 = packed convert + 8B LDS stores.  No divergent control flow.
    v4f xt[8];
    #pragma unroll
    for (int i = 0; i < 8; ++i) {
      int idx = i * 128 + tid;                 // 1024 chunks of 4 floats
      int r = idx >> 3, c4 = (idx & 7) * 4;
      xt[i] = *(const v4f*)&xrow[(size_t)r * kD + k0 + c4];
    }
    // stage W transposed: Ws[n][k] = Wk[h][k0+k][sect*64+n]
    v4f wt[4];
    #pragma unroll
    for (int i = 0; i < 4; ++i) {
      int idx = i * 128 + tid;                 // 512 chunks of 4 floats
      int k = idx >> 4, n4 = (idx & 15) * 4;
      wt[i] = *(const v4f*)&wrow[(size_t)(k0 + k) * kE + n4];
    }
    if (kk + 1 < kD / 32) {  // uniform-condition prefetch of next K-step
      __builtin_prefetch(&xrow[(size_t)(tid >> 3) * kD + k0 + 32 + (tid & 7) * 4], 0, 3);
      __builtin_prefetch(&wrow[(size_t)(k0 + 32 + (tid >> 4)) * kE + (tid & 15) * 4], 0, 3);
    }
    #pragma unroll
    for (int i = 0; i < 8; ++i) {
      int idx = i * 128 + tid;
      int r = idx >> 3, c4 = (idx & 7) * 4;
      *(v4h*)&Xs[r * 32 + c4] = __builtin_convertvector(xt[i], v4h);
    }
    #pragma unroll
    for (int i = 0; i < 4; ++i) {
      int idx = i * 128 + tid;
      int k = idx >> 4, n4 = (idx & 15) * 4;
      v4h hv = __builtin_convertvector(wt[i], v4h);
      #pragma unroll
      for (int u = 0; u < 4; ++u) Ws[(n4 + u) * 32 + k] = hv[u];
    }
    __syncthreads();
    // A fragments: rows wid*32 + g*16 + lm, 16 contiguous K at kb16
    V16 a[2];
    #pragma unroll
    for (int g = 0; g < 2; ++g) {
      const _Float16* ap = &Xs[(wid * 32 + g * 16 + lm) * 32 + kb16];
      a[g].h[0] = *(const v8h*)ap;
      a[g].h[1] = *(const v8h*)(ap + 8);
    }
    #pragma unroll
    for (int t = 0; t < 4; ++t) {
      // B fragment: col t*16+lm, 16 contiguous K at kb16
      V16 bf;
      const _Float16* bp = &Ws[(t * 16 + lm) * 32 + kb16];
      bf.h[0] = *(const v8h*)bp;
      bf.h[1] = *(const v8h*)(bp + 8);
      #pragma unroll
      for (int g = 0; g < 2; ++g)
        acc[g][t] = wmma_f16(a[g].v, bf.v, acc[g][t]);
    }
    __syncthreads();
  }
  const int Mb = (lane < 16) ? 0 : 8;
  _Float16* dst = kqvh + (size_t)sect * kB * kH * kN * kDH;
  #pragma unroll
  for (int t = 0; t < 4; ++t) {
    const int ew = t * 16 + lm;  // 0..63 within the section
    const float bias = bk[h * kE + sect * 64 + ew];
    #pragma unroll
    for (int g = 0; g < 2; ++g)
      #pragma unroll
      for (int r = 0; r < 8; ++r) {
        int n = mbase + wid * 32 + g * 16 + r + Mb;
        dst[((size_t)bh * kN + n) * kDH + ew] = (_Float16)(acc[g][t][r] + bias);
      }
  }
}

// ---------------------------------------------------------------------------
// Kernel 2: flash attention.  Reference computes S = k·q^T (k rows attend),
// causal tril over (row,col), softmax over cols, O = P·v.  One wave handles a
// 16-row n-tile; streams 32-wide m-blocks up to the diagonal with online
// softmax.  V tiles are bulk-copied to LDS with async-to-LDS ops (ASYNCcnt);
// P goes through per-wave LDS to convert C-layout -> A-layout.
// ---------------------------------------------------------------------------
__global__ __launch_bounds__(128) void attn_kernel(
    const _Float16* __restrict__ kqvh, _Float16* __restrict__ sa) {
  __shared__ __align__(16) _Float16 Ps[4][16 * 32];
#if USE_ASYNC
  __shared__ __align__(16) _Float16 Vb[4][32 * kDH];  // per-wave V tile
#endif
  const size_t S = (size_t)kB * kH * kN * kDH;
  const int tid = threadIdx.x;
  const int wid = tid >> 5, lane = tid & 31, lm = lane & 15;
  const int kb8 = (lane >> 4) * 8, kb16 = (lane >> 4) * 16;
  const int widx = blockIdx.x * 4 + wid;
  const int bh = widx >> 7;        // 128 n-tiles per (b,h)
  const int n0 = (widx & 127) * 16;
  const int b = bh / kH, h = bh % kH;
  const _Float16* kp = kqvh + (size_t)bh * kN * kDH;  // "query" role
  const _Float16* qp = kp + S;                        // "key" role
  const _Float16* vp = kp + 2 * S;
  const int Mb = (lane < 16) ? 0 : 8;

  // A fragments of the 16 k-rows, dh split into two K=32 halves
  V16 qa[2];
  #pragma unroll
  for (int hf = 0; hf < 2; ++hf) {
    const _Float16* p = kp + (n0 + lm) * kDH + hf * 32 + kb16;
    qa[hf].h[0] = *(const v8h*)p;
    qa[hf].h[1] = *(const v8h*)(p + 8);
  }

  v8f O[4] = {};
  float mrow[8], lrow[8];
  #pragma unroll
  for (int r = 0; r < 8; ++r) { mrow[r] = -3.0e38f; lrow[r] = 0.f; }

  const int jmax = (n0 + 15) >> 5;
  for (int j = 0; j <= jmax; ++j) {
    const int m0 = j * 32;
#if USE_ASYNC
    // async bulk copy of V tile [m0..m0+31] x 64 dh into per-wave LDS
    {
      _Float16* vb = &Vb[wid][0];
      #pragma unroll
      for (int c = 0; c < 8; ++c) {
        int chunk = c * 32 + lane;      // 256 chunks of 16B (8 f16)
        int rloc = chunk >> 3;          // local m row 0..31
        int coff = (chunk & 7) * 8;     // f16 offset within row
        int grow = m0 + rloc;
        if (grow >= kN) grow = kN - 1;  // clamped rows multiply P==0
        __builtin_amdgcn_global_load_async_to_lds_b128(
            AS1(vp + (size_t)grow * kDH + coff), AS3(vb + rloc * kDH + coff),
            0, 0);
      }
    }
#endif
    v8f s[2] = {};
    #pragma unroll
    for (int sub = 0; sub < 2; ++sub) {
      int col = m0 + sub * 16 + lm;
      if (col >= kN) col = kN - 1;  // masked later; clamp the load
      #pragma unroll
      for (int hf = 0; hf < 2; ++hf) {
        V16 bf;
        const _Float16* p = qp + col * kDH + hf * 32 + kb16;
        bf.h[0] = *(const v8h*)p;
        bf.h[1] = *(const v8h*)(p + 8);
        s[sub] = wmma_f16(qa[hf].v, bf.v, s[sub]);
      }
    }
    // causal mask + 1/sqrt(dh) scale + per-row max
    float rmax[8];
    #pragma unroll
    for (int r = 0; r < 8; ++r) {
      const int row = n0 + r + Mb;
      float v0 = (m0 + lm <= row) ? s[0][r] * 0.125f : -3.0e38f;
      float v1 = (m0 + 16 + lm <= row) ? s[1][r] * 0.125f : -3.0e38f;
      s[0][r] = v0; s[1][r] = v1;
      rmax[r] = fmaxf(v0, v1);
    }
    #pragma unroll
    for (int msk = 1; msk <= 8; msk <<= 1)
      #pragma unroll
      for (int r = 0; r < 8; ++r)
        rmax[r] = fmaxf(rmax[r], __shfl_xor(rmax[r], msk, 32));
    float scale[8], rsum[8];
    #pragma unroll
    for (int r = 0; r < 8; ++r) {
      float mn = fmaxf(mrow[r], rmax[r]);
      scale[r] = __expf(mrow[r] - mn);
      mrow[r] = mn;
      float p0 = __expf(s[0][r] - mn);
      float p1 = __expf(s[1][r] - mn);
      s[0][r] = p0; s[1][r] = p1;
      rsum[r] = p0 + p1;
    }
    #pragma unroll
    for (int msk = 1; msk <= 8; msk <<= 1)
      #pragma unroll
      for (int r = 0; r < 8; ++r)
        rsum[r] += __shfl_xor(rsum[r], msk, 32);
    #pragma unroll
    for (int r = 0; r < 8; ++r) lrow[r] = lrow[r] * scale[r] + rsum[r];
    #pragma unroll
    for (int t = 0; t < 4; ++t)
      #pragma unroll
      for (int r = 0; r < 8; ++r) O[t][r] *= scale[r];
    // P (C-layout) -> LDS -> reload as A fragment (same-wave LDS is in-order)
    _Float16* pb = &Ps[wid][0];
    #pragma unroll
    for (int sub = 0; sub < 2; ++sub)
      #pragma unroll
      for (int r = 0; r < 8; ++r)
        pb[(r + Mb) * 32 + sub * 16 + lm] = (_Float16)s[sub][r];
    asm volatile("" ::: "memory");
    V16 pa;
    const _Float16* pp = pb + lm * 32 + kb16;
    pa.h[0] = *(const v8h*)pp;
    pa.h[1] = *(const v8h*)(pp + 8);
    // V as B fragments (K = 32 m-rows, 4 dh column tiles)
#if USE_ASYNC
    __builtin_amdgcn_s_wait_asynccnt(0);
    const _Float16* vb = &Vb[wid][0];
    const int kloc = (lane < 16) ? 0 : 16;
    #pragma unroll
    for (int t = 0; t < 4; ++t) {
      V16 vf;
      #pragma unroll
      for (int e = 0; e < 16; ++e)
        vf.v[e] = vb[(kloc + e) * kDH + t * 16 + lm];
      O[t] = wmma_f16(pa.v, vf.v, O[t]);
    }
#else
    const int kbase = m0 + ((lane < 16) ? 0 : 16);
    #pragma unroll
    for (int t = 0; t < 4; ++t) {
      V16 vf;
      #pragma unroll
      for (int e = 0; e < 16; ++e) {
        int mr = kbase + e;
        if (mr >= kN) mr = kN - 1;  // corresponding P entries are zero
        vf.v[e] = vp[(size_t)mr * kDH + t * 16 + lm];
      }
      O[t] = wmma_f16(pa.v, vf.v, O[t]);
    }
#endif
  }
  #pragma unroll
  for (int r = 0; r < 8; ++r) lrow[r] = 1.0f / lrow[r];
  #pragma unroll
  for (int t = 0; t < 4; ++t)
    #pragma unroll
    for (int r = 0; r < 8; ++r) {
      const int row = n0 + r + Mb;
      sa[((size_t)b * kN + row) * kD + h * kDH + t * 16 + lm] =
          (_Float16)(O[t][r] * lrow[r]);
    }
}

// ---------------------------------------------------------------------------
// Kernel 3: output projection  out = sa @ Wo + bo   (M=B*N=4096, N=K=1024)
// Same structure as kernel 1; the f16 sa tile is staged with async-to-LDS.
// ---------------------------------------------------------------------------
__global__ __launch_bounds__(128) void proj_kernel(
    const _Float16* __restrict__ sa, const float* __restrict__ Wo,
    const float* __restrict__ bo, float* __restrict__ out) {
  __shared__ __align__(16) _Float16 Xs[128 * 32];
  __shared__ __align__(16) _Float16 Ws[64 * 32];  // transposed [n][k]
  const int tid = threadIdx.x;
  const int wid = tid >> 5, lane = tid & 31, lm = lane & 15;
  const int kb8 = (lane >> 4) * 8, kb16 = (lane >> 4) * 16;
  const int mbase = blockIdx.x * 128;  // rows of flattened [B*N]
  const int e0 = blockIdx.y * 64;      // output columns

  v8f acc[2][4] = {};
  for (int kk = 0; kk < kD / 32; ++kk) {
    const int k0 = kk * 32;
    // stage sa tile 128x32 f16: direct global->LDS async copy (no VGPR trip)
#if USE_ASYNC
    #pragma unroll
    for (int i = 0; i < 4; ++i) {
      int idx = i * 128 + tid;                 // 512 chunks of 16B (8 f16)
      int r = idx >> 2, c8 = (idx & 3) * 8;
      __builtin_amdgcn_global_load_async_to_lds_b128(
          AS1(&sa[(size_t)(mbase + r) * kD + k0 + c8]), AS3(&Xs[r * 32 + c8]),
          0, 0);
    }
#else
    #pragma unroll
    for (int i = 0; i < 4; ++i) {
      int idx = i * 128 + tid;
      int r = idx >> 2, c8 = (idx & 3) * 8;
      *(v8h*)&Xs[r * 32 + c8] = *(const v8h*)&sa[(size_t)(mbase + r) * kD + k0 + c8];
    }
#endif
    // stage Wo transposed (f32 -> f16): clause of float4 loads, then stores
    v4f wt[4];
    #pragma unroll
    for (int i = 0; i < 4; ++i) {
      int idx = i * 128 + tid;                 // 512 chunks of 4 floats
      int k = idx >> 4, n4 = (idx & 15) * 4;
      wt[i] = *(const v4f*)&Wo[(size_t)(k0 + k) * kD + e0 + n4];
    }
    if (kk + 1 < kD / 32) {  // uniform-condition prefetch of next K-step
      __builtin_prefetch(&Wo[(size_t)(k0 + 32 + (tid >> 4)) * kD + e0 + (tid & 15) * 4], 0, 3);
    }
    #pragma unroll
    for (int i = 0; i < 4; ++i) {
      int idx = i * 128 + tid;
      int k = idx >> 4, n4 = (idx & 15) * 4;
      v4h hv = __builtin_convertvector(wt[i], v4h);
      #pragma unroll
      for (int u = 0; u < 4; ++u) Ws[(n4 + u) * 32 + k] = hv[u];
    }
#if USE_ASYNC
    __builtin_amdgcn_s_wait_asynccnt(0);
#endif
    __syncthreads();
    V16 a[2];
    #pragma unroll
    for (int g = 0; g < 2; ++g) {
      const _Float16* ap = &Xs[(wid * 32 + g * 16 + lm) * 32 + kb16];
      a[g].h[0] = *(const v8h*)ap;
      a[g].h[1] = *(const v8h*)(ap + 8);
    }
    #pragma unroll
    for (int t = 0; t < 4; ++t) {
      V16 bf;
      const _Float16* bp = &Ws[(t * 16 + lm) * 32 + kb16];
      bf.h[0] = *(const v8h*)bp;
      bf.h[1] = *(const v8h*)(bp + 8);
      #pragma unroll
      for (int g = 0; g < 2; ++g)
        acc[g][t] = wmma_f16(a[g].v, bf.v, acc[g][t]);
    }
    __syncthreads();
  }
  const int Mb = (lane < 16) ? 0 : 8;
  #pragma unroll
  for (int t = 0; t < 4; ++t) {
    const float bias = bo[e0 + t * 16 + lm];
    #pragma unroll
    for (int g = 0; g < 2; ++g)
      #pragma unroll
      for (int r = 0; r < 8; ++r) {
        int row = mbase + wid * 32 + g * 16 + r + Mb;
        out[(size_t)row * kD + e0 + t * 16 + lm] = acc[g][t][r] + bias;
      }
  }
}

extern "C" void kernel_launch(void* const* d_in, const int* in_sizes, int n_in,
                              void* d_out, int out_size, void* d_ws,
                              size_t ws_size, hipStream_t stream) {
  (void)in_sizes; (void)n_in; (void)out_size; (void)ws_size;
  const float* x    = (const float*)d_in[0];
  const float* Wkqv = (const float*)d_in[1];
  const float* bkqv = (const float*)d_in[2];
  const float* Wo   = (const float*)d_in[3];
  const float* bo   = (const float*)d_in[4];
  float* out = (float*)d_out;

  _Float16* ws = (_Float16*)d_ws;
  const size_t S = (size_t)kB * kH * kN * kDH;  // 4,194,304 elems per section
  _Float16* kqvh = ws;           // [3][B*H][N][DH] f16  (~25 MB)
  _Float16* sa   = ws + 3 * S;   // [B][N][D] f16        (~8.4 MB)

  dim3 g1(kN / 128, 3, kB * kH);
  kqv_kernel<<<g1, 128, 0, stream>>>(x, Wkqv, bkqv, kqvh);

  const int nwaves = kB * kH * (kN / 16);  // 4096 wave-tiles
  attn_kernel<<<dim3(nwaves / 4), 128, 0, stream>>>(kqvh, sa);

  dim3 g3((kB * kN) / 128, kD / 64);
  proj_kernel<<<g3, 128, 0, stream>>>(sa, Wo, bo, out);
}
